// NeuralCDE_68066641707641
// MI455X (gfx1250) — matrix operand order repaired and checked by reading, not compiled, not run on hardware
//
#include <hip/hip_runtime.h>

// Persistent Neural-CDE RK4 kernel for gfx1250 (MI455X).
// Grid: 64 blocks x 512 threads (16 wave32). Each block owns 16 batch rows and
// runs the full 127-step RK4 scan with workgroup-local barriers only.
// All GEMMs via v_wmma_f32_16x16x32_f16; W2^T fragments (4 M-tiles/wave) and
// the b2 bias fragments live in VGPRs (<256 VGPRs -> no s_set_vgpr_msb).

typedef __attribute__((ext_vector_type(16))) _Float16 v16h;
typedef __attribute__((ext_vector_type(8)))  float    v8f;

namespace {

constexpr int B_TOT = 1024;
constexpr int T_PTS = 128;
constexpr int IN_C  = 16;
constexpr int HIDN  = 64;
constexpr int OUT_C = 10;
constexpr int NSTEP = T_PTS - 1;          // 127 intervals
constexpr int CSTR  = NSTEP * IN_C;       // coeff row stride (127*16)
constexpr int NWAVE = 16;                 // waves per block
constexpr int TPW   = 4;                  // big-GEMM M-tiles per wave (64/16)

union HFrag { v16h v; unsigned u[8]; };

__device__ __forceinline__ unsigned pack2(float a, float b) {
  union { _Float16 h[2]; unsigned u; } t;
  t.h[0] = (_Float16)a; t.h[1] = (_Float16)b;
  return t.u;
}

__device__ __forceinline__ float tanh_fast(float x) {
  // tanh(x) = 1 - 2/(exp(2x)+1); v_exp_f32 + fast rcp, saturates correctly
  float e = __expf(2.0f * x);
  return 1.0f - 2.0f / (e + 1.0f);
}

// 16-bit A fragment (16x32, M x K): lane holds row m=lane&15;
// dwords 0..3 -> K pairs {0,2,4,6}+8*half ; dwords 4..7 -> {16,18,20,22}+8*half
__device__ __forceinline__ int a_k0(int c, int d, int half) {
  return 32 * c + 8 * half + ((d < 4) ? 2 * d : 16 + 2 * (d - 4));
}
// 16-bit B fragment (32x16, K x N): lane holds col n=lane&15;
// dwords 0..3 -> K pairs {0,2,4,6}+8*half ; dwords 4..7 -> {16,18,20,22}+8*half
__device__ __forceinline__ int b_k0(int c, int d, int half) {
  return 32 * c + 8 * half + ((d < 4) ? 2 * d : 16 + 2 * (d - 4));
}

__device__ __forceinline__ v8f wmma_f16(const HFrag& a, const HFrag& b, v8f c) {
  return __builtin_amdgcn_wmma_f32_16x16x32_f16(false, a.v, false, b.v,
                                                (short)0, c, false, false);
}

__launch_bounds__(512, 1)
__global__ void cde_rk4_kernel(const float* __restrict__ times,
                               const float* __restrict__ ca,
                               const float* __restrict__ cbp,
                               const float* __restrict__ c2p,
                               const float* __restrict__ c3p,
                               const float* __restrict__ Winit,
                               const float* __restrict__ binit,
                               const float* __restrict__ W1,
                               const float* __restrict__ b1,
                               const float* __restrict__ W2,
                               const float* __restrict__ b2,
                               const float* __restrict__ Wlin,
                               const float* __restrict__ blin,
                               float* __restrict__ out)
{
  // f16 staging buffers, b-major so fragment K-pairs are dword-contiguous
  __shared__ _Float16 zT[16 * HIDN];     // z^T  : [b][h]
  __shared__ _Float16 hT[16 * HIDN];     // h^T  : [b][h]
  __shared__ float    kbuf[16 * HIDN];   // k    : [b][h]

  const int tid  = threadIdx.x;
  const int wave = tid >> 5;
  const int lane = tid & 31;
  const int half = lane >> 4;
  const int n    = lane & 15;            // B/C-fragment column (batch-in-tile) or A row
  const int rowbase = blockIdx.x * 16;   // global batch row base for this block

  // ---- one-time: W2^T A-fragments + b2 bias fragments into VGPRs
  //      wave w owns M-tiles 4w..4w+3 of the 1024-row W2^T
  HFrag w2f[TPW][2];
  float b2f[TPW][8];
#pragma unroll
  for (int p = 0; p < TPW; ++p) {
    const int t = wave * TPW + p;
    const int m = 16 * t + n;                  // y-column (h*16+i)
#pragma unroll
    for (int c = 0; c < 2; ++c)
#pragma unroll
      for (int d = 0; d < 8; ++d) {
        const int k0 = a_k0(c, d, half);       // hidden index
        w2f[p][c].u[d] = pack2(W2[(size_t)k0 * 1024 + m],
                               W2[(size_t)(k0 + 1) * 1024 + m]);
      }
#pragma unroll
    for (int d = 0; d < 8; ++d)
      b2f[p][d] = b2[16 * t + d + 8 * half];
  }

  // ---- role weights (branch-free loads: clamp address, select value)
  HFrag w1f[2];  float b1f[8];                 // waves 0-3: W1^T tile + bias frag
  HFrag wif;     float bif[8];                 // waves 0-3: W_init^T (K=16, padded)
  HFrag wlf[2];  float blf[8];                 // wave 4   : W_lin^T (rows 10..15 zero)
  if (wave < 4) {
    const int m = 16 * wave + n;               // h_out
#pragma unroll
    for (int c = 0; c < 2; ++c)
#pragma unroll
      for (int d = 0; d < 8; ++d) {
        const int k0 = a_k0(c, d, half);
        w1f[c].u[d] = pack2(W1[k0 * HIDN + m], W1[(k0 + 1) * HIDN + m]);
      }
#pragma unroll
    for (int d = 0; d < 8; ++d) {
      const int k0 = a_k0(0, d, half);
      const int ks = (k0 < IN_C - 1) ? k0 : (IN_C - 2);
      const unsigned wv = pack2(Winit[ks * HIDN + m], Winit[(ks + 1) * HIDN + m]);
      wif.u[d] = (d < 4) ? wv : 0u;            // K >= 16 zero-padded
      b1f[d] = b1[16 * wave + d + 8 * half];
      bif[d] = binit[16 * wave + d + 8 * half];
    }
  } else if (wave == 4) {
    const int o_s = (n < OUT_C) ? n : (OUT_C - 1);
#pragma unroll
    for (int c = 0; c < 2; ++c)
#pragma unroll
      for (int d = 0; d < 8; ++d) {
        const int k0 = a_k0(c, d, half);
        const unsigned wv = pack2(Wlin[k0 * OUT_C + o_s],
                                  Wlin[(k0 + 1) * OUT_C + o_s]);
        wlf[c].u[d] = (n < OUT_C) ? wv : 0u;
      }
#pragma unroll
    for (int d = 0; d < 8; ++d) {
      const int o = d + 8 * half;
      blf[d] = blin[(o < OUT_C) ? o : 0];
      blf[d] = (o < OUT_C) ? blf[d] : 0.0f;
    }
  }

  // ---- stage-store helper (waves 0-3 only): z-stage C tile -> zT f16
  auto stage_store = [&](const float* zs) {
    unsigned* zrow = (unsigned*)&zT[n * HIDN + 16 * wave + 8 * half];
#pragma unroll
    for (int d = 0; d < 4; ++d) zrow[d] = pack2(zs[2 * d], zs[2 * d + 1]);
  };

  // ---- z0 = coeff_a[:,0,:] @ W_init + b_init  (one WMMA per z-tile, K 16->32)
  v8f zacc = {};                                // waves 0-3: persistent z state
  if (wave < 4) {
    HFrag a0f;
    const float* arow = ca + (size_t)(rowbase + n) * CSTR;  // coeff_a[b,0,:]
#pragma unroll
    for (int d = 0; d < 8; ++d) {
      const int k0 = b_k0(0, d, half);          // spline channel i
      const int ks = (k0 < IN_C - 1) ? k0 : 0;  // d>=4 (k>=16) -> padded zeros
      const unsigned av = pack2(arow[ks], arow[ks + 1]);
      a0f.u[d] = (k0 + 1 < IN_C) ? av : 0u;
    }
    v8f c0 = {};
    zacc = wmma_f16(wif, a0f, c0);
#pragma unroll
    for (int d = 0; d < 8; ++d) zacc[d] += bif[d];
    float zs[8];
#pragma unroll
    for (int d = 0; d < 8; ++d) zs[d] = zacc[d];
    stage_store(zs);
  }
  __syncthreads();

  // ---- output projection (wave 4): out[:, ti, :] = W_lin^T @ z^T + b_lin
  auto out_proj = [&](int ti) {
    if (wave == 4) {
      HFrag zb[2];
#pragma unroll
      for (int c = 0; c < 2; ++c)
#pragma unroll
        for (int d = 0; d < 8; ++d)
          zb[c].u[d] = *(const unsigned*)&zT[n * HIDN + b_k0(c, d, half)];
      v8f acc = {};
      acc = wmma_f16(wlf[0], zb[0], acc);
      acc = wmma_f16(wlf[1], zb[1], acc);
#pragma unroll
      for (int d = 0; d < 8; ++d) {
        const int o = d + 8 * half;
        if (o < OUT_C)
          out[((size_t)(rowbase + n) * T_PTS + ti) * OUT_C + o] = acc[d] + blf[d];
      }
    }
  };
  out_proj(0);

  // ---- F(z, dx) -> k   (z staged in zT; dx fragment: dx[b=n][i=d+8*half])
  auto F_eval = [&](const float* dx, float* kout) {
    if (wave < 4) {                             // h^T tile w = relu(W1^T @ z^T + b1)
      HFrag zb[2];
#pragma unroll
      for (int c = 0; c < 2; ++c)
#pragma unroll
        for (int d = 0; d < 8; ++d)
          zb[c].u[d] = *(const unsigned*)&zT[n * HIDN + b_k0(c, d, half)];
      v8f hacc = {};
      hacc = wmma_f16(w1f[0], zb[0], hacc);
      hacc = wmma_f16(w1f[1], zb[1], hacc);
      unsigned* hrow = (unsigned*)&hT[n * HIDN + 16 * wave + 8 * half];
#pragma unroll
      for (int d = 0; d < 4; ++d)
        hrow[d] = pack2(fmaxf(hacc[2 * d]     + b1f[2 * d],     0.0f),
                        fmaxf(hacc[2 * d + 1] + b1f[2 * d + 1], 0.0f));
    }
    __syncthreads();                            // h^T ready

    // big GEMM Y^T = W2^T @ h^T per tile, fused bias + tanh + einsum reduction
    HFrag hb[2];
#pragma unroll
    for (int c = 0; c < 2; ++c)
#pragma unroll
      for (int d = 0; d < 8; ++d)
        hb[c].u[d] = *(const unsigned*)&hT[n * HIDN + b_k0(c, d, half)];
#pragma unroll
    for (int p = 0; p < TPW; ++p) {
      const int t = wave * TPW + p;             // output hidden unit h = t
      v8f y = {};
      y = wmma_f16(w2f[p][0], hb[0], y);
      y = wmma_f16(w2f[p][1], hb[1], y);
      float acc = 0.0f;
#pragma unroll
      for (int d = 0; d < 8; ++d) {             // i = d + 8*half along VGPR dim
        const float f = tanh_fast(y[d] + b2f[p][d]);
        acc = fmaf(f, dx[d], acc);
      }
      acc += __shfl_xor(acc, 16);               // combine i-halves across lane halves
      if (half == 0) kbuf[n * HIDN + t] = acc;  // k[b][h]
    }
    __syncthreads();                            // kbuf ready

    if (wave < 4) {
#pragma unroll
      for (int d = 0; d < 8; ++d)
        kout[d] = kbuf[n * HIDN + 16 * wave + 8 * half + d];
    }
  };

  // ---- spline derivative state
  float cbv[8], c2v[8], c3v[8];
  float g1[8];                                  // dx for k1 (carried g4 of prev step)
  {
    const float* p0 = cbp + (size_t)(rowbase + n) * CSTR + 8 * half; // interval 0, frac 0
#pragma unroll
    for (int d = 0; d < 8; ++d) g1[d] = p0[d];
  }

  float k1[8], k2[8], k3[8], k4[8], zs[8], g[8];

  // ================= RK4 (3/8 rule) scan over 127 intervals =================
  for (int i = 0; i < NSTEP; ++i) {
    const float t0 = times[i];
    const float dt = times[i + 1] - t0;
    const size_t off = ((size_t)(rowbase + n) * NSTEP + i) * IN_C + 8 * half;
#pragma unroll
    for (int d = 0; d < 8; ++d) {
      cbv[d] = cbp[off + d]; c2v[d] = c2p[off + d]; c3v[d] = c3p[off + d];
    }
    if (i + 1 < NSTEP) {                        // prefetch next interval's coeffs
      const size_t nx = off + IN_C;
      __builtin_prefetch(&cbp[nx], 0, 1);
      __builtin_prefetch(&c2p[nx], 0, 1);
      __builtin_prefetch(&c3p[nx], 0, 1);
    }
    auto eval_dx = [&](float f, float* gd) {
#pragma unroll
      for (int d = 0; d < 8; ++d)
        gd[d] = fmaf(fmaf(c3v[d], f, c2v[d]), f, cbv[d]);
    };

    // stage 1: k1 = F(z, dx(t0));  z1 = z + dt*k1/3
    F_eval(g1, k1);
    if (wave < 4) {
      const float s = dt * (1.0f / 3.0f);
#pragma unroll
      for (int d = 0; d < 8; ++d) zs[d] = fmaf(s, k1[d], zacc[d]);
      stage_store(zs);
    }
    __syncthreads();

    // stage 2: k2 = F(z1, dx(t0+dt/3));  z2 = z + dt*(k2 - k1/3)
    eval_dx(dt * (1.0f / 3.0f), g);
    F_eval(g, k2);
    if (wave < 4) {
#pragma unroll
      for (int d = 0; d < 8; ++d)
        zs[d] = fmaf(dt, k2[d] - k1[d] * (1.0f / 3.0f), zacc[d]);
      stage_store(zs);
    }
    __syncthreads();

    // stage 3: k3 = F(z2, dx(t0+2dt/3));  z3 = z + dt*(k1 - k2 + k3)
    eval_dx(dt * (2.0f / 3.0f), g);
    F_eval(g, k3);
    if (wave < 4) {
#pragma unroll
      for (int d = 0; d < 8; ++d)
        zs[d] = fmaf(dt, k1[d] - k2[d] + k3[d], zacc[d]);
      stage_store(zs);
    }
    __syncthreads();

    // stage 4: k4 = F(z3, dx(t1));  zn = z + dt*(k1 + 3(k2+k3) + k4)/8
    eval_dx(dt, g1);                            // g1 becomes next step's k1 deriv
    F_eval(g1, k4);
    if (wave < 4) {
#pragma unroll
      for (int d = 0; d < 8; ++d) {
        zacc[d] = fmaf(dt * 0.125f,
                       k1[d] + 3.0f * (k2[d] + k3[d]) + k4[d], zacc[d]);
        zs[d] = zacc[d];
      }
      stage_store(zs);
    }
    __syncthreads();

    out_proj(i + 1);
  }
}

} // namespace

extern "C" void kernel_launch(void* const* d_in, const int* in_sizes, int n_in,
                              void* d_out, int out_size, void* d_ws, size_t ws_size,
                              hipStream_t stream) {
  (void)in_sizes; (void)n_in; (void)d_ws; (void)ws_size; (void)out_size;
  const float* times = (const float*)d_in[0];
  const float* ca    = (const float*)d_in[1];
  const float* cbp   = (const float*)d_in[2];
  const float* c2p   = (const float*)d_in[3];
  const float* c3p   = (const float*)d_in[4];
  // d_in[5] = final_index (unused in stream=True mode)
  const float* Winit = (const float*)d_in[6];
  const float* binit = (const float*)d_in[7];
  const float* W1    = (const float*)d_in[8];
  const float* b1    = (const float*)d_in[9];
  const float* W2    = (const float*)d_in[10];
  const float* b2    = (const float*)d_in[11];
  const float* Wlin  = (const float*)d_in[12];
  const float* blin  = (const float*)d_in[13];
  float* out = (float*)d_out;

  cde_rk4_kernel<<<dim3(B_TOT / 16), dim3(512), 0, stream>>>(
      times, ca, cbp, c2p, c3p, Winit, binit, W1, b1, W2, b2, Wlin, blin, out);
}
